// MLAFlashAttention_7112465842672
// MI455X (gfx1250) — matrix operand order repaired and checked by reading, not compiled, not run on hardware
//
#include <hip/hip_runtime.h>
#include <hip/hip_bf16.h>
#include <cstddef>

// ---------------- MLA constants (match reference) ----------------
#define SEQ      2048
#define HIDDEN   2048
#define NHEADS   16
#define QLORA    1536
#define KVLORA   512
#define NOPE     128
#define ROPE     64
#define VDIM     128
#define QKDIM    192            // NOPE + ROPE
#define SEG      1024           // cu_seqlens = [0,1024,2048]
#define SCALE    0.07216878364870323f   // 1/sqrt(192)
#define EPS      1e-6f

typedef __bf16 bf16;
typedef __attribute__((ext_vector_type(16))) __bf16 v16bf;
typedef __attribute__((ext_vector_type(8)))  __bf16 v8bf;
typedef __attribute__((ext_vector_type(8)))  float  v8f;

// Assemble a 16-element bf16 fragment from two 16-byte chunks.
__device__ inline v16bf ldfrag(const bf16* p0, const bf16* p1) {
    v8bf a = *reinterpret_cast<const v8bf*>(p0);
    v8bf b = *reinterpret_cast<const v8bf*>(p1);
    return __builtin_shufflevector(a, b, 0,1,2,3,4,5,6,7,8,9,10,11,12,13,14,15);
}

__device__ inline v8f wmma_bf16(v16bf a, v16bf b, v8f c) {
    return __builtin_amdgcn_wmma_f32_16x16x32_bf16(false, a, false, b, (short)0, c,
                                                   false, false);
}

// ---------------- elementwise converters ----------------
__global__ void cvt_bf16(const float* __restrict__ in, bf16* __restrict__ out, size_t n) {
    size_t i = (size_t)blockIdx.x * blockDim.x + threadIdx.x;
    if (i < n) out[i] = (bf16)in[i];
}

// in[R][C] (row-major) -> out[C][R]  (i.e. store B transposed: Bt[n*K+k])
__global__ void cvt_bf16_T(const float* __restrict__ in, bf16* __restrict__ out,
                           int R, int C) {
    size_t i = (size_t)blockIdx.x * blockDim.x + threadIdx.x;
    if (i < (size_t)R * C) {
        int r = (int)(i / C), c = (int)(i % C);
        out[(size_t)c * R + r] = (bf16)in[i];
    }
}

// ---------------- RMSNorm (block per row) ----------------
__global__ __launch_bounds__(256) void rmsnorm_bf16(const float* __restrict__ in,
                                                    int rstride, int D,
                                                    const float* __restrict__ w,
                                                    bf16* __restrict__ out) {
    __shared__ float red[8];
    int row = blockIdx.x;
    const float* x = in + (size_t)row * rstride;
    float ss = 0.f;
    for (int i = threadIdx.x; i < D; i += blockDim.x) { float v = x[i]; ss += v * v; }
    #pragma unroll
    for (int off = 1; off < 32; off <<= 1) ss += __shfl_xor(ss, off, 32);
    int wave = threadIdx.x >> 5, lane = threadIdx.x & 31;
    if (lane == 0) red[wave] = ss;
    __syncthreads();
    float tot = 0.f;
    int nw = blockDim.x >> 5;
    for (int i = 0; i < nw; ++i) tot += red[i];
    float rs = rsqrtf(tot / (float)D + EPS);
    for (int i = threadIdx.x; i < D; i += blockDim.x)
        out[(size_t)row * D + i] = (bf16)(x[i] * rs * w[i]);
}

// ---------------- generic bf16 WMMA GEMM ----------------
// C[M][N] (f32, row-major) = A[M][K] (bf16, row-major) * Bt[N][K] (bf16, B transposed)
// One wave computes a 64x64 tile: 4x4 wmma tiles, K stepped by 32.
__global__ __launch_bounds__(256, 1) void gemm_bf16_wmma(const bf16* __restrict__ A,
                                                         const bf16* __restrict__ Bt,
                                                         float* __restrict__ C,
                                                         int M, int N, int K) {
    int lane  = threadIdx.x & 31;
    int wave  = threadIdx.x >> 5;
    int tilesN = N >> 6;
    int tile  = blockIdx.x * (blockDim.x >> 5) + wave;
    int total = (M >> 6) * tilesN;
    if (tile >= total) return;
    int tm = (tile / tilesN) << 6;
    int tn = (tile % tilesN) << 6;
    int row = lane & 15, khalf = lane >> 4;

    v8f acc[4][4] = {};
    for (int k = 0; k < K; k += 32) {
        v16bf af[4], bfv[4];
        #pragma unroll
        for (int mi = 0; mi < 4; ++mi) {
            // ISA 16-bit A layout: lane half selects K sub-block (khalf*8, +16)
            const bf16* pa = A + (size_t)(tm + mi * 16 + row) * K + k + khalf * 8;
            af[mi] = ldfrag(pa, pa + 16);
        }
        #pragma unroll
        for (int nj = 0; nj < 4; ++nj) {
            // B layout: same K order as A (khalf*8, +16)
            const bf16* pb = Bt + (size_t)(tn + nj * 16 + row) * K + k + khalf * 8;
            bfv[nj] = ldfrag(pb, pb + 16);
        }
        #pragma unroll
        for (int mi = 0; mi < 4; ++mi)
            #pragma unroll
            for (int nj = 0; nj < 4; ++nj)
                acc[mi][nj] = wmma_bf16(af[mi], bfv[nj], acc[mi][nj]);
    }
    // C layout: vgpr r holds row khalf*8+r, col = lane%16 -> coalesced f32 stores
    #pragma unroll
    for (int mi = 0; mi < 4; ++mi)
        #pragma unroll
        for (int nj = 0; nj < 4; ++nj) {
            int r0 = tm + mi * 16 + khalf * 8;
            int c0 = tn + nj * 16 + row;
            #pragma unroll
            for (int r = 0; r < 8; ++r)
                C[(size_t)(r0 + r) * N + c0] = acc[mi][nj][r];
        }
}

// ---------------- RoPE + head assembly ----------------
// qf  : [SEQ][NHEADS*192]   (Q up-proj output)
// kvf : [SEQ][NHEADS*256]   (KV up-proj output: 128 k_nope + 128 v per head)
// ckv : [SEQ][576]          (KV down-proj output: 512 kv_c + 64 k_rope)
// Qb/Kb : [NHEADS][SEQ][192] bf16 ; Vt : [NHEADS][VDIM][SEQ] bf16 (transposed)
__global__ __launch_bounds__(128) void rope_assemble(const float* __restrict__ qf,
                                                     const float* __restrict__ kvf,
                                                     const float* __restrict__ ckv,
                                                     const float* __restrict__ cosp,
                                                     const float* __restrict__ sinp,
                                                     bf16* __restrict__ Qb,
                                                     bf16* __restrict__ Kb,
                                                     bf16* __restrict__ Vt) {
    int s = blockIdx.x, h = blockIdx.y, t = threadIdx.x;
    size_t qo = ((size_t)h * SEQ + s) * QKDIM;
    const float* qrow  = qf  + (size_t)s * (NHEADS * QKDIM) + h * QKDIM;
    const float* kvrow = kvf + (size_t)s * (NHEADS * (NOPE + VDIM)) + h * (NOPE + VDIM);
    // nope halves + V (transposed store)
    Qb[qo + t] = (bf16)qrow[t];
    Kb[qo + t] = (bf16)kvrow[t];
    Vt[((size_t)h * VDIM + t) * SEQ + s] = (bf16)kvrow[NOPE + t];
    if (t < 32) {
        float c  = cosp[(size_t)s * ROPE + t];
        float sn = sinp[(size_t)s * ROPE + t];
        float q0 = qrow[NOPE + 2 * t], q1 = qrow[NOPE + 2 * t + 1];
        Qb[qo + NOPE + t]      = (bf16)(q0 * c - q1 * sn);
        Qb[qo + NOPE + 32 + t] = (bf16)(q1 * c + q0 * sn);
        const float* kr = ckv + (size_t)s * (KVLORA + ROPE) + KVLORA;
        float k0 = kr[2 * t], k1 = kr[2 * t + 1];
        Kb[qo + NOPE + t]      = (bf16)(k0 * c - k1 * sn);
        Kb[qo + NOPE + 32 + t] = (bf16)(k1 * c + k0 * sn);
    }
}

// ---------------- flash attention (segment-causal) ----------------
// grid = (SEQ/64, NHEADS), block = 128 (4 waves). Wave w owns 16 query rows.
// Emits the normalized output directly as bf16 (A-matrix of the final GEMM),
// avoiding an f32 round-trip + conversion pass.
__global__ __launch_bounds__(128, 1) void mla_flash(const bf16* __restrict__ Qb,
                                                    const bf16* __restrict__ Kb,
                                                    const bf16* __restrict__ Vt,
                                                    bf16* __restrict__ Oout) {
    __shared__ bf16 plds[4][16 * 64];   // per-wave P tile (16 q rows x 64 keys)
    int lane = threadIdx.x & 31;
    int wave = threadIdx.x >> 5;
    int h = blockIdx.y;
    int qbase = blockIdx.x * 64;
    int row = lane & 15, khalf = lane >> 4;
    const bf16* Qh = Qb + (size_t)h * SEQ * QKDIM;
    const bf16* Kh = Kb + (size_t)h * SEQ * QKDIM;
    const bf16* Vh = Vt + (size_t)h * VDIM * SEQ;
    int qrow0 = qbase + wave * 16;

    // preload Q fragments (6 x K=32 along d=192)
    v16bf qf[6];
    #pragma unroll
    for (int ds = 0; ds < 6; ++ds) {
        const bf16* p = Qh + (size_t)(qrow0 + row) * QKDIM + ds * 32 + khalf * 8;
        qf[ds] = ldfrag(p, p + 16);
    }

    float m_i[8], l_i[8];
    v8f Oacc[8] = {};
    #pragma unroll
    for (int r = 0; r < 8; ++r) { m_i[r] = -1e30f; l_i[r] = 0.f; }

    int seg0 = (qbase / SEG) * SEG;
    for (int kb = seg0; kb < qbase + 64; kb += 64) {
        // ---- S = Q K^T for 16 q-rows x 64 keys ----
        v8f s[4] = {};
        #pragma unroll
        for (int ds = 0; ds < 6; ++ds) {
            #pragma unroll
            for (int nb = 0; nb < 4; ++nb) {
                const bf16* p = Kh + (size_t)(kb + nb * 16 + row) * QKDIM + ds * 32 + khalf * 8;
                v16bf kf = ldfrag(p, p + 16);
                s[nb] = wmma_bf16(qf[ds], kf, s[nb]);
            }
        }
        // ---- scale + causal mask + per-row block max ----
        float bm[8];
        #pragma unroll
        for (int r = 0; r < 8; ++r) bm[r] = -1e30f;
        int qb0 = qrow0 + khalf * 8;
        #pragma unroll
        for (int nb = 0; nb < 4; ++nb) {
            int key = kb + nb * 16 + row;
            #pragma unroll
            for (int r = 0; r < 8; ++r) {
                float v = s[nb][r] * SCALE;
                v = (key <= qb0 + r) ? v : -1e30f;
                s[nb][r] = v;
                bm[r] = fmaxf(bm[r], v);
            }
        }
        #pragma unroll
        for (int r = 0; r < 8; ++r) {   // row lives in one VGPR across a 16-lane half
            float v = bm[r];
            #pragma unroll
            for (int off = 1; off < 16; off <<= 1) v = fmaxf(v, __shfl_xor(v, off, 32));
            bm[r] = v;
        }
        // ---- online softmax update ----
        float alpha[8], rsum[8];
        #pragma unroll
        for (int r = 0; r < 8; ++r) {
            float mn = fmaxf(m_i[r], bm[r]);
            alpha[r] = __expf(m_i[r] - mn);
            m_i[r] = mn;
            rsum[r] = 0.f;
        }
        #pragma unroll
        for (int nb = 0; nb < 4; ++nb)
            #pragma unroll
            for (int r = 0; r < 8; ++r) {
                float pv = __expf(s[nb][r] - m_i[r]);
                s[nb][r] = pv;
                rsum[r] += pv;
            }
        #pragma unroll
        for (int r = 0; r < 8; ++r) {
            float v = rsum[r];
            #pragma unroll
            for (int off = 1; off < 16; off <<= 1) v += __shfl_xor(v, off, 32);
            l_i[r] = l_i[r] * alpha[r] + v;
        }
        // ---- re-layout P via LDS (C layout -> A layout) ----
        __syncthreads();
        #pragma unroll
        for (int nb = 0; nb < 4; ++nb)
            #pragma unroll
            for (int r = 0; r < 8; ++r)
                plds[wave][(khalf * 8 + r) * 64 + nb * 16 + row] = (bf16)s[nb][r];
        __syncthreads();
        // ---- O = O*alpha + P V ----
        #pragma unroll
        for (int nt = 0; nt < 8; ++nt) {
            v8f o = Oacc[nt];
            #pragma unroll
            for (int r = 0; r < 8; ++r) o[r] *= alpha[r];
            Oacc[nt] = o;
        }
        #pragma unroll
        for (int kk = 0; kk < 2; ++kk) {
            const bf16* pp = &plds[wave][row * 64 + kk * 32 + khalf * 8];
            v16bf pf = ldfrag(pp, pp + 16);
            #pragma unroll
            for (int nt = 0; nt < 8; ++nt) {
                const bf16* pv = Vh + (size_t)(nt * 16 + row) * SEQ + kb + kk * 32 + khalf * 8;
                v16bf vf = ldfrag(pv, pv + 16);
                Oacc[nt] = wmma_bf16(pf, vf, Oacc[nt]);
            }
        }
    }
    // ---- epilogue: normalize, convert, store attn_bf16[q][h*128 + vd] ----
    #pragma unroll
    for (int nt = 0; nt < 8; ++nt) {
        float inv;
        #pragma unroll
        for (int r = 0; r < 8; ++r) {
            inv = 1.0f / l_i[r];
            int q = qrow0 + khalf * 8 + r;
            Oout[(size_t)q * (NHEADS * VDIM) + h * VDIM + nt * 16 + row] =
                (bf16)(Oacc[nt][r] * inv);
        }
    }
}

// ---------------- host orchestration ----------------
static inline int gemm_blocks(int M, int N) {
    int tiles = (M / 64) * (N / 64);
    return (tiles + 7) / 8;   // 8 waves (256 threads) per block
}

extern "C" void kernel_launch(void* const* d_in, const int* in_sizes, int n_in,
                              void* d_out, int out_size, void* d_ws, size_t ws_size,
                              hipStream_t stream) {
    (void)in_sizes; (void)n_in; (void)out_size; (void)ws_size;
    const float* hidden = (const float*)d_in[0];
    const float* cosp   = (const float*)d_in[1];
    const float* sinp   = (const float*)d_in[2];
    const float* wq_a   = (const float*)d_in[3];
    const float* q_nw   = (const float*)d_in[4];
    const float* wq_b   = (const float*)d_in[5];
    const float* wkv_a  = (const float*)d_in[6];
    const float* kv_nw  = (const float*)d_in[7];
    const float* wkv_b  = (const float*)d_in[8];
    const float* wo     = (const float*)d_in[9];
    float* out = (float*)d_out;

    char* p = (char*)d_ws;
    auto alloc = [&](size_t bytes) -> void* {
        void* r = (void*)p;
        p += (bytes + 255) & ~(size_t)255;
        return r;
    };
    bf16*  hb      = (bf16*) alloc((size_t)SEQ * HIDDEN * 2);
    bf16*  wqa_t   = (bf16*) alloc((size_t)QLORA * HIDDEN * 2);
    bf16*  wkva_t  = (bf16*) alloc((size_t)(KVLORA + ROPE) * HIDDEN * 2);
    float* qlora   = (float*)alloc((size_t)SEQ * QLORA * 4);
    float* ckv     = (float*)alloc((size_t)SEQ * (KVLORA + ROPE) * 4);
    bf16*  qn      = (bf16*) alloc((size_t)SEQ * QLORA * 2);
    bf16*  kvn     = (bf16*) alloc((size_t)SEQ * KVLORA * 2);
    bf16*  wqb_t   = (bf16*) alloc((size_t)(NHEADS * QKDIM) * QLORA * 2);
    bf16*  wkvb_t  = (bf16*) alloc((size_t)(NHEADS * (NOPE + VDIM)) * KVLORA * 2);
    float* qfull   = (float*)alloc((size_t)SEQ * NHEADS * QKDIM * 4);
    float* kvfull  = (float*)alloc((size_t)SEQ * NHEADS * (NOPE + VDIM) * 4);
    bf16*  Qb      = (bf16*) alloc((size_t)NHEADS * SEQ * QKDIM * 2);
    bf16*  Kb      = (bf16*) alloc((size_t)NHEADS * SEQ * QKDIM * 2);
    bf16*  Vt      = (bf16*) alloc((size_t)NHEADS * VDIM * SEQ * 2);
    bf16*  attn_b  = (bf16*) alloc((size_t)SEQ * NHEADS * VDIM * 2);
    bf16*  wo_t    = (bf16*) alloc((size_t)HIDDEN * NHEADS * VDIM * 2);

    // 1) convert activations + weights to bf16 (weights transposed to Bt layout)
    {
        size_t n = (size_t)SEQ * HIDDEN;
        cvt_bf16<<<(unsigned)((n + 255) / 256), 256, 0, stream>>>(hidden, hb, n);
    }
    {
        size_t n = (size_t)HIDDEN * QLORA;
        cvt_bf16_T<<<(unsigned)((n + 255) / 256), 256, 0, stream>>>(wq_a, wqa_t, HIDDEN, QLORA);
    }
    {
        size_t n = (size_t)HIDDEN * (KVLORA + ROPE);
        cvt_bf16_T<<<(unsigned)((n + 255) / 256), 256, 0, stream>>>(wkv_a, wkva_t, HIDDEN, KVLORA + ROPE);
    }
    // 2) down projections
    gemm_bf16_wmma<<<gemm_blocks(SEQ, QLORA), 256, 0, stream>>>(hb, wqa_t, qlora, SEQ, QLORA, HIDDEN);
    gemm_bf16_wmma<<<gemm_blocks(SEQ, KVLORA + ROPE), 256, 0, stream>>>(hb, wkva_t, ckv, SEQ, KVLORA + ROPE, HIDDEN);
    // 3) RMSNorm
    rmsnorm_bf16<<<SEQ, 256, 0, stream>>>(qlora, QLORA, QLORA, q_nw, qn);
    rmsnorm_bf16<<<SEQ, 256, 0, stream>>>(ckv, KVLORA + ROPE, KVLORA, kv_nw, kvn);
    // 4) up projections
    {
        size_t n = (size_t)QLORA * NHEADS * QKDIM;
        cvt_bf16_T<<<(unsigned)((n + 255) / 256), 256, 0, stream>>>(wq_b, wqb_t, QLORA, NHEADS * QKDIM);
    }
    {
        size_t n = (size_t)KVLORA * NHEADS * (NOPE + VDIM);
        cvt_bf16_T<<<(unsigned)((n + 255) / 256), 256, 0, stream>>>(wkv_b, wkvb_t, KVLORA, NHEADS * (NOPE + VDIM));
    }
    gemm_bf16_wmma<<<gemm_blocks(SEQ, NHEADS * QKDIM), 256, 0, stream>>>(qn, wqb_t, qfull, SEQ, NHEADS * QKDIM, QLORA);
    gemm_bf16_wmma<<<gemm_blocks(SEQ, NHEADS * (NOPE + VDIM)), 256, 0, stream>>>(kvn, wkvb_t, kvfull, SEQ, NHEADS * (NOPE + VDIM), KVLORA);
    // 5) RoPE + per-head Q/K/V assembly (V transposed for PV WMMA B-frags)
    {
        dim3 g(SEQ, NHEADS);
        rope_assemble<<<g, 128, 0, stream>>>(qfull, kvfull, ckv, cosp, sinp, Qb, Kb, Vt);
    }
    // 6) flash attention -> bf16 attention output (A matrix of final GEMM)
    {
        dim3 g(SEQ / 64, NHEADS);
        mla_flash<<<g, 128, 0, stream>>>(Qb, Kb, Vt, attn_b);
    }
    // 7) output projection -> d_out
    {
        size_t n = (size_t)(NHEADS * VDIM) * HIDDEN;
        cvt_bf16_T<<<(unsigned)((n + 255) / 256), 256, 0, stream>>>(wo, wo_t, NHEADS * VDIM, HIDDEN);
    }
    gemm_bf16_wmma<<<gemm_blocks(SEQ, HIDDEN), 256, 0, stream>>>(attn_b, wo_t, out, SEQ, HIDDEN, NHEADS * VDIM);
}
